// Separable_Conv2Plus1D_43997644981013
// MI455X (gfx1250) — matrix operand order repaired and checked by reading, not compiled, not run on hardware
//
#include <hip/hip_runtime.h>

// ---------------------------------------------------------------------------
// Separable (2+1)D conv for MI455X (gfx1250, wave32, WMMA).
//   x[8,16,56,56,64] --dw3x3/VALID--> --pw 64->128 +b--> y[8,54,54,16,128] (bf16, in d_ws)
//   y --conv1d over t, KT=3, 128->128 +b--> out[8,14,54,54,128] (f32)
// Matmuls run on v_wmma_f32_16x16x32_bf16 (f32 accumulate).
// ---------------------------------------------------------------------------

typedef __attribute__((ext_vector_type(16))) __bf16 v16bf;
typedef __attribute__((ext_vector_type(2)))  __bf16 v2bf;
typedef __attribute__((ext_vector_type(8)))  float  v8f;
typedef __attribute__((ext_vector_type(4)))  float  v4f;
typedef __attribute__((ext_vector_type(4)))  unsigned int v4u;

#if defined(__has_builtin)
#  if __has_builtin(__builtin_amdgcn_cvt_pk_bf16_f32)
#    define HAVE_CVT_PK_BF16 1
#  endif
#endif
#ifndef HAVE_CVT_PK_BF16
#  define HAVE_CVT_PK_BF16 0
#endif

namespace {
constexpr int NB = 8, NT = 16, NH = 56, NW = 56, NC = 64, NF = 128;
constexpr int HP = 54, WP = 54, TP = 14;
constexpr int PIX = HP * WP;                 // 2916
constexpr int M1  = NB * NT * PIX;           // 373248 spatial GEMM rows
constexpr int M2  = NB * PIX * TP;           // 326592 temporal GEMM rows
constexpr int TILES1 = M1 / 16;              // 23328 (exact)
constexpr int TILES2 = (M2 + 15) / 16;       // 20412 (exact)
}

// pack two f32 into one bf16x2 dword (lo = first arg); native v_cvt_pk_bf16_f32
__device__ __forceinline__ unsigned pack2(float lo, float hi) {
#if HAVE_CVT_PK_BF16
    auto p = __builtin_amdgcn_cvt_pk_bf16_f32(lo, hi);
    static_assert(sizeof(p) == 4, "bf16x2 expected");
    unsigned r; __builtin_memcpy(&r, &p, 4);
    return r;
#else
    v2bf p;
    p.x = (__bf16)lo;
    p.y = (__bf16)hi;
    unsigned r; __builtin_memcpy(&r, &p, 4);
    return r;
#endif
}

__device__ __forceinline__ unsigned short bf_bits(float x) {
    __bf16 b = (__bf16)x;
    return __builtin_bit_cast(unsigned short, b);
}

// wave32 lane broadcast via ds_bpermute_b32 (lane-permute HW, no LDS memory)
__device__ __forceinline__ int lane_bcast(int val, int srcLane) {
    return __builtin_amdgcn_ds_bpermute(srcLane << 2, val);
}

// ===========================================================================
// Kernel 1: fused depthwise 3x3 + pointwise 64->128 (+bias), bf16 out to ws.
// 256 threads = 8 waves; each wave owns one 16-row (pixel) x 128-col tile.
// ===========================================================================
__global__ __launch_bounds__(256) void k_spatial(
    const float* __restrict__ x,  const float* __restrict__ dw,
    const float* __restrict__ pw, const float* __restrict__ b2d,
    unsigned short* __restrict__ y)
{
    // pointwise weights pre-swizzled to exact B-fragment layout:
    //   [frag = kc*8+nt][lane][e]  where k = kc*32 + (lane>>4)*8 + e + (e>=8 ? 8 : 0), f = nt*16 + (lane&15)
    __shared__ unsigned short pwArr[16 * 32 * 16];   // 16 KB (bf16 bits)
    __shared__ float dwLds[9 * NC];                  // 2.25 KB
    __shared__ float b2dLds[NF];

    const int tid = threadIdx.x;
    for (int i = tid; i < 9 * NC; i += 256) dwLds[i] = dw[i];
    for (int i = tid; i < NF;     i += 256) b2dLds[i] = b2d[i];
    // fill as packed bf16x2 words: word w = elements {2w, 2w+1} (e even, e+1 -> k+1)
    unsigned* pwW = (unsigned*)pwArr;
    for (int w = tid; w < 16 * 32 * 16 / 2; w += 256) {
        int i    = 2 * w;
        int e    = i & 15;                // even
        int lane = (i >> 4) & 31;
        int frag = i >> 9;                // 0..15
        int nt = frag & 7, kc = frag >> 3;
        int hi = lane >> 4, n = lane & 15;
        int k = kc * 32 + hi * 8 + e + (e >= 8 ? 8 : 0);
        int f = nt * 16 + n;
        pwW[w] = pack2(pw[k * NF + f], pw[(k + 1) * NF + f]);
    }
    __syncthreads();

    const int lane = tid & 31;
    const int wave = tid >> 5;
    const int hi   = lane >> 4;
    const int nc   = lane & 15;                    // A: row within tile; B/C: column
    const int mBase = (blockIdx.x * 8 + wave) * 16;

    // ---- decompose own row once (feeds both A loads and the store offsets) ----
    const int m   = mBase + nc;
    const int bt  = m / PIX;
    const int rem = m % PIX;
    const int oh  = rem / WP;
    const int ow  = rem % WP;
    const float* xbase = x + (((long)bt * NH + oh) * NW + ow) * NC;

    // own row's output offset in y[b][oh][ow][t][f]; rows j of the tile live on
    // lanes j (=0..15), so fan out the 8 rows of this half via lane permute.
    const int bb = bt >> 4, tt = bt & 15;          // bt = b*16 + t
    const int myOff = (((bb * HP + oh) * WP + ow) * NT + tt) * NF;
    int rowOff[8];
    #pragma unroll
    for (int r = 0; r < 8; ++r)
        rowOff[r] = lane_bcast(myOff, hi * 8 + r);

    // ---- build both 16x32 bf16 A-fragments (depthwise output) in registers ----
    v16bf afr[2];
    #pragma unroll
    for (int kc = 0; kc < 2; ++kc) {
        const int c0 = kc * 32 + hi * 8;          // channels for e = 0..7
        const int c1 = c0 + 16;                   // channels for e = 8..15
        v4f a0 = {0.f, 0.f, 0.f, 0.f}, a1 = a0, a2 = a0, a3 = a0;
        #pragma unroll
        for (int kh = 0; kh < 3; ++kh) {
            #pragma unroll
            for (int kw = 0; kw < 3; ++kw) {
                const float* px = xbase + (kh * NW + kw) * NC;
                const float* wv = &dwLds[(kh * 3 + kw) * NC];
                a0 += *(const v4f*)(px + c0)     * *(const v4f*)(wv + c0);
                a1 += *(const v4f*)(px + c0 + 4) * *(const v4f*)(wv + c0 + 4);
                a2 += *(const v4f*)(px + c1)     * *(const v4f*)(wv + c1);
                a3 += *(const v4f*)(px + c1 + 4) * *(const v4f*)(wv + c1 + 4);
            }
        }
        union { v16bf v; unsigned w[8]; } u;
        u.w[0] = pack2(a0.x, a0.y); u.w[1] = pack2(a0.z, a0.w);
        u.w[2] = pack2(a1.x, a1.y); u.w[3] = pack2(a1.z, a1.w);
        u.w[4] = pack2(a2.x, a2.y); u.w[5] = pack2(a2.z, a2.w);
        u.w[6] = pack2(a3.x, a3.y); u.w[7] = pack2(a3.z, a3.w);
        afr[kc] = u.v;
    }

    // ---- 8 N-tiles x (2 WMMAs of K=32) ----
    #pragma unroll
    for (int nt = 0; nt < 8; ++nt) {
        const float bias = b2dLds[nt * 16 + nc];
        v8f acc = {bias, bias, bias, bias, bias, bias, bias, bias};
        #pragma unroll
        for (int kc = 0; kc < 2; ++kc) {
            const v16bf bfr = *(const v16bf*)&pwArr[((kc * 8 + nt) * 32 + lane) * 16];
            acc = __builtin_amdgcn_wmma_f32_16x16x32_bf16(
                      false, afr[kc], false, bfr, (short)0, acc, false, false);
        }
        #pragma unroll
        for (int r = 0; r < 8; ++r)
            y[rowOff[r] + nt * 16 + nc] = bf_bits(acc[r]);
    }
}

// ===========================================================================
// Kernel 2: temporal conv1d as GEMM, K = 3*128 via 12 accumulating WMMAs.
// grid.y halves the 128 filters so static LDS (B-fragments) stays at 48 KB.
// ===========================================================================
__global__ __launch_bounds__(256) void k_temporal(
    const unsigned short* __restrict__ y, const float* __restrict__ k1d,
    const float* __restrict__ b1d, float* __restrict__ out)
{
    // k1d pre-swizzled B-fragments: [frag = (kt*4+kc)*4+ntl][lane][e]
    __shared__ unsigned short kArr[48 * 32 * 16];   // 48 KB (bf16 bits)
    __shared__ float b1Lds[64];

    const int tid   = threadIdx.x;
    const int fBase = blockIdx.y * 64;

    for (int i = tid; i < 64; i += 256) b1Lds[i] = b1d[fBase + i];
    unsigned* kW = (unsigned*)kArr;
    for (int w = tid; w < 48 * 32 * 16 / 2; w += 256) {
        int i    = 2 * w;
        int e    = i & 15;                 // even
        int lane = (i >> 4) & 31;
        int frag = i >> 9;                 // 0..47
        int ntl = frag & 3, kc = (frag >> 2) & 3, kt = frag >> 4;
        int hi = lane >> 4, n = lane & 15;
        int fin = kc * 32 + hi * 8 + e + (e >= 8 ? 8 : 0);
        int f   = fBase + ntl * 16 + n;
        kW[w] = pack2(k1d[(kt * NF + fin) * NF + f],
                      k1d[(kt * NF + fin + 1) * NF + f]);
    }
    __syncthreads();

    const int lane = tid & 31;
    const int wave = tid >> 5;
    const int tile = blockIdx.x * 8 + wave;
    if (tile >= TILES2) return;           // wave-uniform: EXEC stays all-ones
    const int hi = lane >> 4;
    const int nc = lane & 15;
    const int mBase = tile * 16;

    // ---- decompose own row once ----
    const int m   = mBase + nc;
    const int pix = m / TP;
    const int tp  = m % TP;
    const unsigned short* ybase = y + ((long)pix * NT + tp) * NF;

    // own row's offset in out[b][tp][oh][ow][f], fan out via lane permute
    const int bb  = pix / PIX;
    const int rr  = pix % PIX;
    const int ohr = rr / WP;
    const int owr = rr % WP;
    const int myOff = (((bb * TP + tp) * HP + ohr) * WP + owr) * NF + fBase;
    int rowOff[8];
    #pragma unroll
    for (int r = 0; r < 8; ++r)
        rowOff[r] = lane_bcast(myOff, hi * 8 + r);

    // ---- preload all 12 A-fragments (bf16 vector loads from ws) ----
    v16bf afr[12];
    #pragma unroll
    for (int kt = 0; kt < 3; ++kt) {
        #pragma unroll
        for (int kc = 0; kc < 4; ++kc) {
            union { v16bf v; v4u q[2]; } u;
            const unsigned short* p = ybase + kt * NF + kc * 32 + hi * 8;
            u.q[0] = *(const v4u*)(p);        // e=0..7  : fin = kc*32 + hi*8 + e
            u.q[1] = *(const v4u*)(p + 16);   // e=8..15 : fin = kc*32 + 16 + hi*8 + (e-8)
            afr[kt * 4 + kc] = u.v;
        }
    }

    // ---- 4 N-tiles x (3 taps x 4 K-chunks) = 12 accumulating WMMAs each ----
    #pragma unroll
    for (int ntl = 0; ntl < 4; ++ntl) {
        const float bias = b1Lds[ntl * 16 + nc];
        v8f acc = {bias, bias, bias, bias, bias, bias, bias, bias};
        #pragma unroll
        for (int kt = 0; kt < 3; ++kt) {
            #pragma unroll
            for (int kc = 0; kc < 4; ++kc) {
                const v16bf bfr =
                    *(const v16bf*)&kArr[(((kt * 4 + kc) * 4 + ntl) * 32 + lane) * 16];
                acc = __builtin_amdgcn_wmma_f32_16x16x32_bf16(
                          false, afr[kt * 4 + kc], false, bfr, (short)0, acc, false, false);
            }
        }
        #pragma unroll
        for (int r = 0; r < 8; ++r)
            out[rowOff[r] + ntl * 16 + nc] = acc[r];
    }
}

// ===========================================================================
extern "C" void kernel_launch(void* const* d_in, const int* in_sizes, int n_in,
                              void* d_out, int out_size, void* d_ws, size_t ws_size,
                              hipStream_t stream) {
    const float* x   = (const float*)d_in[0];
    const float* dw  = (const float*)d_in[1];
    const float* pw  = (const float*)d_in[2];
    const float* b2d = (const float*)d_in[3];
    const float* k1d = (const float*)d_in[4];
    const float* b1d = (const float*)d_in[5];
    float* out = (float*)d_out;

    // intermediate y: bf16 [8][54][54][16][128] = 95,551,488 bytes in d_ws
    unsigned short* ybuf = (unsigned short*)d_ws;
    (void)in_sizes; (void)n_in; (void)out_size; (void)ws_size;

    k_spatial<<<TILES1 / 8, 256, 0, stream>>>(x, dw, pw, b2d, ybuf);
    dim3 g2((TILES2 + 7) / 8, 2, 1);
    k_temporal<<<g2, 256, 0, stream>>>(ybuf, k1d, b1d, out);
}
